// RWKV_TimeMix_69295002353973
// MI455X (gfx1250) — matrix operand order repaired and checked by reading, not compiled, not run on hardware
//
#include <hip/hip_runtime.h>
#include <hip/hip_bf16.h>

// Problem constants (match reference)
#define BB   4
#define TT   2048
#define CC   1024
#define HH   16
#define DD   64
#define NROW (BB * TT)          // 8192 rows for all GEMMs

typedef __bf16  v16bf  __attribute__((ext_vector_type(16)));
typedef __bf16  bf16x8 __attribute__((ext_vector_type(8)));
typedef float   v8f    __attribute__((ext_vector_type(8)));

// ---------------------------------------------------------------------------
// fp32 -> bf16 convert (weights), float4 vectorized
// ---------------------------------------------------------------------------
__global__ void cvt_f32_bf16(const float* __restrict__ src,
                             __bf16* __restrict__ dst, int n) {
    int i = (blockIdx.x * blockDim.x + threadIdx.x) * 4;
    if (i + 3 < n) {
        float4 f = *(const float4*)(src + i);
        dst[i + 0] = (__bf16)f.x;
        dst[i + 1] = (__bf16)f.y;
        dst[i + 2] = (__bf16)f.z;
        dst[i + 3] = (__bf16)f.w;
    }
}

// ---------------------------------------------------------------------------
// Token-shift mixing: xk/xv/xr = x*mix + x_prev*(1-mix), bf16 out, 4-wide
// ---------------------------------------------------------------------------
__global__ void mix_kernel(const float* __restrict__ x,
                           const float* __restrict__ mk,
                           const float* __restrict__ mv,
                           const float* __restrict__ mr,
                           __bf16* __restrict__ xk,
                           __bf16* __restrict__ xv,
                           __bf16* __restrict__ xr) {
    int i = (blockIdx.x * blockDim.x + threadIdx.x) * 4;   // over NROW*CC
    int c = i & (CC - 1);
    int n = i >> 10;               // row = b*T + t  (CC == 1024)
    int t = n & (TT - 1);
    float4 xc = *(const float4*)(x + i);
    float4 xp = (t == 0) ? make_float4(0.f, 0.f, 0.f, 0.f)
                         : *(const float4*)(x + i - CC);
    float4 m;
#define MIX4(MPTR, OPTR)                                                    \
    m = *(const float4*)(MPTR + c);                                         \
    OPTR[i + 0] = (__bf16)(xc.x * m.x + xp.x * (1.0f - m.x));               \
    OPTR[i + 1] = (__bf16)(xc.y * m.y + xp.y * (1.0f - m.y));               \
    OPTR[i + 2] = (__bf16)(xc.z * m.z + xp.z * (1.0f - m.z));               \
    OPTR[i + 3] = (__bf16)(xc.w * m.w + xp.w * (1.0f - m.w));
    MIX4(mk, xk)
    MIX4(mv, xv)
    MIX4(mr, xr)
#undef MIX4
}

// ---------------------------------------------------------------------------
// WMMA bf16 GEMM:  C[m,n] = sum_k A[m,k] * W[n,k]   (out = A @ W^T), fp32 out
// Per wave: one 32(M) x 64(N) tile -> 8 WMMAs per K-step on 12 b128 loads.
// A operand k order chosen here (matches B): lane L<16 -> row M=L, K=0..15 contiguous;
//                                     lane L>=16 -> row M=L-16, K=16..31.
// B operand k order chosen here (matches A): lanes 0..15 -> N=lane, K=0..15 contiguous;
//                                     lanes 16..31 -> N=lane-16, K=16..31.
// ---------------------------------------------------------------------------
__device__ __forceinline__ v16bf load_a_frag(const __bf16* p) {
    bf16x8 c0 = *(const bf16x8*)(p);
    bf16x8 c1 = *(const bf16x8*)(p + 8);
    return __builtin_shufflevector(c0, c1, 0, 1, 2, 3, 4, 5, 6, 7,
                                   8, 9, 10, 11, 12, 13, 14, 15);
}

#define MT   32                  // M-tile per wave
#define NT   64                  // N-tile per wave
#define WPB  8                   // waves per block

__global__ void __launch_bounds__(256)
gemm_bf16_tn(const __bf16* __restrict__ A,     // M x K, row-major
             const __bf16* __restrict__ W,     // N x K, row-major
             float* __restrict__ Cout,         // M x N
             int M, int N, int K) {
    const int lane = threadIdx.x & 31;
    const int wave = threadIdx.x >> 5;
    const int m0 = (blockIdx.x * WPB + wave) * MT;
    const int n0 = blockIdx.y * NT;
    const int lr  = lane & 15;
    const int hiA = (lane >> 4) * 16;   // A K-sub-block offset
    const int hiB = (lane >> 4) * 16;   // B K-sub-block offset

    const __bf16* pa = A + (size_t)(m0 + lr) * K + hiA;
    const __bf16* pb = W + (size_t)(n0 + lr) * K + hiB;
    const size_t rowK16 = (size_t)16 * K;

    v8f acc00 = {}, acc01 = {}, acc02 = {}, acc03 = {};
    v8f acc10 = {}, acc11 = {}, acc12 = {}, acc13 = {};

    for (int k0 = 0; k0 < K; k0 += 32) {
        __builtin_prefetch(pa + k0 + 32, 0, 1);        // global_prefetch_b8
        v16bf a0 = load_a_frag(pa + k0);
        v16bf a1 = load_a_frag(pa + rowK16 + k0);
        v16bf b0 = *(const v16bf*)(pb + k0);
        v16bf b1 = *(const v16bf*)(pb + 1 * rowK16 + k0);
        v16bf b2 = *(const v16bf*)(pb + 2 * rowK16 + k0);
        v16bf b3 = *(const v16bf*)(pb + 3 * rowK16 + k0);
        acc00 = __builtin_amdgcn_wmma_f32_16x16x32_bf16(false, a0, false, b0,
                                                        (short)0, acc00, false, false);
        acc01 = __builtin_amdgcn_wmma_f32_16x16x32_bf16(false, a0, false, b1,
                                                        (short)0, acc01, false, false);
        acc02 = __builtin_amdgcn_wmma_f32_16x16x32_bf16(false, a0, false, b2,
                                                        (short)0, acc02, false, false);
        acc03 = __builtin_amdgcn_wmma_f32_16x16x32_bf16(false, a0, false, b3,
                                                        (short)0, acc03, false, false);
        acc10 = __builtin_amdgcn_wmma_f32_16x16x32_bf16(false, a1, false, b0,
                                                        (short)0, acc10, false, false);
        acc11 = __builtin_amdgcn_wmma_f32_16x16x32_bf16(false, a1, false, b1,
                                                        (short)0, acc11, false, false);
        acc12 = __builtin_amdgcn_wmma_f32_16x16x32_bf16(false, a1, false, b2,
                                                        (short)0, acc12, false, false);
        acc13 = __builtin_amdgcn_wmma_f32_16x16x32_bf16(false, a1, false, b3,
                                                        (short)0, acc13, false, false);
    }

    // C/D layout: VGPR i, lanes 0-15 -> (M=m0+i, N=n0+lane),
    //             lanes 16-31 -> (M=m0+8+i, N=n0+lane-16)
    const int mo = (lane >> 4) * 8;
    float* po0 = Cout + (size_t)(m0 + mo) * N + n0 + lr;
    float* po1 = po0 + (size_t)16 * N;
#pragma unroll
    for (int i = 0; i < 8; ++i) {
        po0[(size_t)i * N +  0] = acc00[i];
        po0[(size_t)i * N + 16] = acc01[i];
        po0[(size_t)i * N + 32] = acc02[i];
        po0[(size_t)i * N + 48] = acc03[i];
        po1[(size_t)i * N +  0] = acc10[i];
        po1[(size_t)i * N + 16] = acc11[i];
        po1[(size_t)i * N + 32] = acc12[i];
        po1[(size_t)i * N + 48] = acc13[i];
    }
}

// ---------------------------------------------------------------------------
// WKV recurrence (+ fused sigmoid(r) * wkv, output bf16)
// One wave32 per (b,h); each lane owns 2 channels (D=64).
//   a[t] = alpha*a[t-1] + exp(k[t-1])*v[t-1]            (per channel)
//   b[t] = alpha*b[t-1] + sum_d exp(k[t-1,d])           (per head)
//   wkv  = (a + eu*ek*v) / max(b + eu*sum_d ek, 1e-6)
// k/v/r loads are pipelined one step ahead; exp/reduce/div are off the chain.
// ---------------------------------------------------------------------------
__global__ void __launch_bounds__(32)
wkv_kernel(const float* __restrict__ kf, const float* __restrict__ vf,
           const float* __restrict__ rf,
           const float* __restrict__ time_decay,
           const float* __restrict__ time_first,
           __bf16* __restrict__ out) {
    const int bh = blockIdx.x;
    const int b  = bh / HH;
    const int h  = bh % HH;
    const int lane = threadIdx.x;

    const size_t base = (size_t)b * TT * CC + (size_t)h * DD + lane * 2;
    const float2* kp = (const float2*)(kf + base);
    const float2* vp = (const float2*)(vf + base);
    const float2* rp = (const float2*)(rf + base);
    __bf16* op = out + base;
    const int strideT = CC / 2;   // float2 stride per timestep

    const float alpha = __expf(-__expf(time_decay[h]));
    const float eu    = __expf(time_first[h]);

    float a0 = 0.0f, a1 = 0.0f, bs = 0.0f;
    float2 kk = kp[0], vv = vp[0], rr = rp[0];

    for (int t = 0; t < TT; ++t) {
        float2 nk, nv, nr;
        if (t + 1 < TT) {                 // software-pipelined prefetch
            nk = kp[(size_t)(t + 1) * strideT];
            nv = vp[(size_t)(t + 1) * strideT];
            nr = rp[(size_t)(t + 1) * strideT];
        }
        float ek0 = __expf(kk.x);
        float ek1 = __expf(kk.y);
        float s = ek0 + ek1;              // head-wide sum over D (64 = 32 lanes x 2)
        s += __shfl_xor(s, 16);
        s += __shfl_xor(s, 8);
        s += __shfl_xor(s, 4);
        s += __shfl_xor(s, 2);
        s += __shfl_xor(s, 1);

        float num0 = fmaf(eu * ek0, vv.x, a0);
        float num1 = fmaf(eu * ek1, vv.y, a1);
        float den  = fmaxf(fmaf(eu, s, bs), 1e-6f);
        float inv  = __builtin_amdgcn_rcpf(den);
        float r0   = __builtin_amdgcn_rcpf(1.0f + __expf(-rr.x));
        float r1   = __builtin_amdgcn_rcpf(1.0f + __expf(-rr.y));

        op[(size_t)t * CC + 0] = (__bf16)(r0 * num0 * inv);
        op[(size_t)t * CC + 1] = (__bf16)(r1 * num1 * inv);

        // carried chain: 3 FMAs
        a0 = fmaf(alpha, a0, ek0 * vv.x);
        a1 = fmaf(alpha, a1, ek1 * vv.y);
        bs = fmaf(alpha, bs, s);

        kk = nk; vv = nv; rr = nr;
    }
}

// ---------------------------------------------------------------------------
// Launch
// ---------------------------------------------------------------------------
extern "C" void kernel_launch(void* const* d_in, const int* in_sizes, int n_in,
                              void* d_out, int out_size, void* d_ws, size_t ws_size,
                              hipStream_t stream) {
    const float* x          = (const float*)d_in[0];
    const float* mix_k      = (const float*)d_in[1];
    const float* mix_v      = (const float*)d_in[2];
    const float* mix_r      = (const float*)d_in[3];
    const float* W_key      = (const float*)d_in[4];
    const float* W_value    = (const float*)d_in[5];
    const float* W_recept   = (const float*)d_in[6];
    const float* W_output   = (const float*)d_in[7];
    const float* time_decay = (const float*)d_in[8];
    const float* time_first = (const float*)d_in[9];
    float* out = (float*)d_out;

    char* ws = (char*)d_ws;
    const size_t ACT_BF16 = (size_t)NROW * CC * sizeof(__bf16);   // 16 MB
    const size_t W_BF16   = (size_t)CC * CC * sizeof(__bf16);     // 2 MB
    const size_t ACT_F32  = (size_t)NROW * CC * sizeof(float);    // 32 MB

    __bf16* xk  = (__bf16*)(ws);                                  // reused for r*wkv
    __bf16* xv  = (__bf16*)(ws + ACT_BF16);
    __bf16* xr  = (__bf16*)(ws + 2 * ACT_BF16);
    __bf16* wkb = (__bf16*)(ws + 3 * ACT_BF16);
    __bf16* wvb = (__bf16*)((char*)wkb + W_BF16);
    __bf16* wrb = (__bf16*)((char*)wkb + 2 * W_BF16);
    __bf16* wob = (__bf16*)((char*)wkb + 3 * W_BF16);
    float*  kf  = (float*)(ws + 3 * ACT_BF16 + 4 * W_BF16);
    float*  vf  = (float*)((char*)kf + ACT_F32);
    float*  rf  = (float*)((char*)kf + 2 * ACT_F32);

    // 1) weights -> bf16
    {
        int n = CC * CC;
        int blocks = n / (256 * 4);
        cvt_f32_bf16<<<blocks, 256, 0, stream>>>(W_key,    wkb, n);
        cvt_f32_bf16<<<blocks, 256, 0, stream>>>(W_value,  wvb, n);
        cvt_f32_bf16<<<blocks, 256, 0, stream>>>(W_recept, wrb, n);
        cvt_f32_bf16<<<blocks, 256, 0, stream>>>(W_output, wob, n);
    }

    // 2) token-shift mix -> bf16 activations (4 elements per thread)
    mix_kernel<<<(NROW * CC) / (256 * 4), 256, 0, stream>>>(x, mix_k, mix_v, mix_r,
                                                            xk, xv, xr);

    // 3) k / v / r_pre GEMMs (WMMA bf16 -> f32)
    dim3 ggrid(NROW / (MT * WPB), CC / NT);   // (32, 16)
    gemm_bf16_tn<<<ggrid, 256, 0, stream>>>(xk, wkb, kf, NROW, CC, CC);
    gemm_bf16_tn<<<ggrid, 256, 0, stream>>>(xv, wvb, vf, NROW, CC, CC);
    gemm_bf16_tn<<<ggrid, 256, 0, stream>>>(xr, wrb, rf, NROW, CC, CC);

    // 4) WKV recurrence + sigmoid(r)*wkv -> bf16 (written over xk, now dead)
    __bf16* rwkv = xk;
    wkv_kernel<<<BB * HH, 32, 0, stream>>>(kf, vf, rf, time_decay, time_first, rwkv);

    // 5) final projection GEMM -> fp32 output
    gemm_bf16_tn<<<ggrid, 256, 0, stream>>>(rwkv, wob, out, NROW, CC, CC);
}
